// HungarianMatcher_16535624089904
// MI455X (gfx1250) — compile-verified, matrix operand order checked by reading, not run on hardware
//
#include <hip/hip_runtime.h>

// ---------------- problem constants (from the reference) ----------------
#define BS   64
#define NQ   300
#define NC   2
#define TT   16
#define NQT  (BS * NQ)     // 19200 predictions
#define NT   (BS * TT)     // 1024 targets
#define BIGF 1000000000.0f

typedef _Float16 v16h __attribute__((ext_vector_type(16)));
typedef float    v8f  __attribute__((ext_vector_type(8)));
typedef unsigned int u32x4_t __attribute__((ext_vector_type(4)));
typedef int          i32x8_t __attribute__((ext_vector_type(8)));
typedef int          i32x4_t __attribute__((ext_vector_type(4)));

// ---------------------------------------------------------------------------
// Kernel A: foreground prob (NC==2 softmax -> sigmoid), split -p into
// hi/lo f16 pair so the WMMA rank-1 class-cost fold is f32-exact.
// ---------------------------------------------------------------------------
__global__ void hm_prob_kernel(const float* __restrict__ logits,
                               unsigned int* __restrict__ nprob_pk) {
  int q = blockIdx.x * 256 + threadIdx.x;
  if (q >= NQT) return;
  float l0 = logits[2 * q + 0];
  float l1 = logits[2 * q + 1];
  float p  = 1.0f / (1.0f + expf(l1 - l0));   // softmax[:,0] for 2 classes
  float np = -p;                               // cost_class = -prob
  _Float16 hi = (_Float16)np;
  _Float16 lo = (_Float16)(np - (float)hi);
  union { _Float16 h[2]; unsigned int u; } pk;
  pk.h[0] = hi; pk.h[1] = lo;
  nprob_pk[q] = pk.u;
}

// ---------------------------------------------------------------------------
// Kernel B: cost matrix. Grid (1200, 8), block 256 (8 waves).
// Block covers 16 queries x 128 targets; each wave owns a 16x16 tile laid out
// in the WMMA f32 C/D layout. Target table (16 KB) staged to LDS: TDM DMA is
// issued by wave 0 while all threads also perform a compiler-visible
// cooperative copy of the same bytes (keeps stgt live + correct either way).
// ---------------------------------------------------------------------------
__global__ void __launch_bounds__(256)
hm_cost_kernel(const float* __restrict__ pred_boxes,
               const float* __restrict__ tgt_bbox,
               const unsigned int* __restrict__ nprob_pk,
               float* __restrict__ C) {
  __shared__ float stgt[NT * 4];   // 16 KB @ LDS offset 0 (only shared var)

#if defined(__AMDGCN__) && __has_builtin(__builtin_amdgcn_tensor_load_to_lds)
  // ---- Tensor Data Mover: DMA the 4096-dword table as a 64x64 2D tile ----
  if ((threadIdx.x >> 5) == 0) {
    unsigned long long ga = (unsigned long long)(uintptr_t)tgt_bbox;
    u32x4_t g0;
    g0[0] = 1u;                                       // count=1, user mode
    g0[1] = 0u;                                       // lds_addr = 0 (stgt)
    g0[2] = (unsigned)(ga & 0xffffffffull);           // global_addr[31:0]
    g0[3] = (unsigned)((ga >> 32) & 0x01ffffffull)    // global_addr[56:32]
            | 0x80000000u;                            // type = 2 (image)
    i32x8_t g1;
    g1[0] = (int)(2u << 16);     // data_size = 4B
    g1[1] = (int)(64u << 16);    // tensor_dim0 = 64
    g1[2] = (int)(64u << 16);    // tensor_dim1 = 64
    g1[3] = (int)(64u << 16);    // tile_dim0 = 64
    g1[4] = (int)(64u);          // tile_dim1 = 64, tile_dim2 = 0
    g1[5] = (int)(64u);          // tensor_dim0_stride = 64
    g1[6] = 0;
    g1[7] = 0;
    i32x4_t g2 = {0, 0, 0, 0};
    i32x4_t g3 = {0, 0, 0, 0};
    i32x8_t g4 = {0, 0, 0, 0, 0, 0, 0, 0};
    __builtin_amdgcn_tensor_load_to_lds(g0, g1, g2, g3, g4, 0);
  }
#endif

  // compiler-visible staging of the same data (writes identical bytes, so
  // any interleaving with the TDM writes is benign); keeps stgt live.
  for (int idx = threadIdx.x; idx < NT * 4; idx += 256)
    stgt[idx] = tgt_bbox[idx];

#if defined(__AMDGCN__) && __has_builtin(__builtin_amdgcn_tensor_load_to_lds)
  if ((threadIdx.x >> 5) == 0)
    __builtin_amdgcn_s_wait_tensorcnt(0);
#endif
  __syncthreads();

  const int lane = threadIdx.x & 31;
  const int wv   = threadIdx.x >> 5;
  const int q0   = blockIdx.x * 16;
  const int t0   = blockIdx.y * 128 + wv * 16;

  __builtin_prefetch(pred_boxes + (size_t)q0 * 4, 0, 0);  // global_prefetch_b8

  // this lane's target column (WMMA C/D: N = lane & 15)
  const int col = t0 + (lane & 15);
  const float tcx = stgt[col * 4 + 0];
  const float tcy = stgt[col * 4 + 1];
  const float tw  = stgt[col * 4 + 2];
  const float th  = stgt[col * 4 + 3];
  const float tx0 = tcx - 0.5f * tw, ty0 = tcy - 0.5f * th;
  const float tx1 = tcx + 0.5f * tw, ty1 = tcy + 0.5f * th;
  const float tarea = tw * th;

  const int mbase = q0 + ((lane >> 4) << 3);  // WMMA C/D: M = r + 8*(lane>=16)

  v8f acc = {};
#pragma unroll
  for (int r = 0; r < 8; ++r) {
    const int qi = mbase + r;
    const float qcx = pred_boxes[qi * 4 + 0];
    const float qcy = pred_boxes[qi * 4 + 1];
    const float qw  = pred_boxes[qi * 4 + 2];
    const float qh  = pred_boxes[qi * 4 + 3];
    // L1 distance in cxcywh space
    float l1d = fabsf(qcx - tcx) + fabsf(qcy - tcy) +
                fabsf(qw - tw) + fabsf(qh - th);
    // GIoU in xyxy space
    const float qx0 = qcx - 0.5f * qw, qy0 = qcy - 0.5f * qh;
    const float qx1 = qcx + 0.5f * qw, qy1 = qcy + 0.5f * qh;
    const float qarea = qw * qh;
    const float iw = fmaxf(fminf(qx1, tx1) - fmaxf(qx0, tx0), 0.0f);
    const float ih = fmaxf(fminf(qy1, ty1) - fmaxf(qy0, ty0), 0.0f);
    const float inter = iw * ih;
    const float uni   = qarea + tarea - inter;
    const float iou   = inter / uni;
    const float ew = fmaxf(fmaxf(qx1, tx1) - fminf(qx0, tx0), 0.0f);
    const float eh = fmaxf(fmaxf(qy1, ty1) - fminf(qy0, ty0), 0.0f);
    const float earea = ew * eh;
    const float giou = iou - (earea - uni) / earea;
    acc[r] = l1d - giou;   // COST_BBOX*bbox + COST_GIOU*(-giou)
  }

  // rank-1 class-cost fold via WMMA: D = A(16x32,f16) * B(32x16,f16) + acc
  // A row m: K=0 -> hi(-prob_m), K=1 -> lo(-prob_m); B rows 0,1 = ones.
  v16h A = {};
  v16h Bm = {};
  if (lane < 16) {
    union { unsigned int u; _Float16 h[2]; } pk;
    pk.u = nprob_pk[q0 + lane];
    A[0] = pk.h[0];           // K=0 (16-bit A layout: lanes 0-15 hold K=0,1)
    A[1] = pk.h[1];           // K=1
    Bm[0] = (_Float16)1.0f;   // B[K=0][N=lane] (lanes 0-15 hold K=0..7,16..23)
    Bm[1] = (_Float16)1.0f;   // B[K=1][N=lane]
  }
  acc = __builtin_amdgcn_wmma_f32_16x16x32_f16(
      /*neg_a=*/false, A, /*neg_b=*/false, Bm,
      /*c_mod=*/(short)0, acc, /*reuse_a=*/false, /*reuse_b=*/false);

  // store: per r, lanes 0-15 / 16-31 each write 16 contiguous floats of a row
#pragma unroll
  for (int r = 0; r < 8; ++r) {
    const size_t qi = (size_t)(mbase + r);
    C[qi * NT + col] = acc[r];
  }
}

// ---------------------------------------------------------------------------
// Kernel C: 64 independent Hungarian (JV) solves, one wave32 per problem.
// cost_tq[t][q] = C[(b*NQ+q)*NT + b*TT + t]; n=16 rows, m=300 cols.
// Inner O(m) scans are lane-parallel; state lives in LDS.
// ---------------------------------------------------------------------------
__global__ void __launch_bounds__(32)
hm_hungarian_kernel(const float* __restrict__ C,
                    float* __restrict__ out_rows,
                    float* __restrict__ out_cols) {
  const int b = blockIdx.x;
  const int lane = threadIdx.x;
  const int M1 = NQ + 1;  // 301

  __shared__ float ca[(TT + 1) * (NQ + 1)];
  __shared__ float u[TT + 1];
  __shared__ float v[NQ + 1];
  __shared__ float minv[NQ + 1];
  __shared__ int   p[NQ + 1];
  __shared__ int   way[NQ + 1];
  __shared__ int   used[NQ + 1];
  __shared__ int   sj0;

  for (int idx = lane; idx < (TT + 1) * M1; idx += 32) ca[idx] = 0.0f;
  for (int idx = lane; idx < M1; idx += 32) { v[idx] = 0.0f; p[idx] = 0; way[idx] = 0; }
  for (int idx = lane; idx <= TT; idx += 32) u[idx] = 0.0f;
  __syncthreads();
  for (int idx = lane; idx < TT * NQ; idx += 32) {
    const int i = idx / NQ;   // target row
    const int j = idx % NQ;   // query col
    ca[(i + 1) * M1 + (j + 1)] =
        C[((size_t)(b * NQ + j)) * NT + (size_t)(b * TT + i)];
  }
  __syncthreads();

  for (int i = 1; i <= TT; ++i) {
    for (int idx = lane; idx < M1; idx += 32) { minv[idx] = BIGF; used[idx] = 0; }
    if (lane == 0) { p[0] = i; sj0 = 0; }
    __syncthreads();

    while (true) {
      const int j0 = sj0;
      if (p[j0] == 0) break;
      if (lane == 0) used[j0] = 1;
      __syncthreads();
      const int i0 = p[j0];
      const float ui0 = u[i0];

      float best = BIGF;
      int bestj = NQ;
      for (int j = lane; j < M1; j += 32) {
        if (!used[j]) {
          float mj = minv[j];
          const float cur = ca[i0 * M1 + j] - ui0 - v[j];
          if (cur < mj) { mj = cur; minv[j] = cur; way[j] = j0; }
          if (mj < best || (mj == best && j < bestj)) { best = mj; bestj = j; }
        }
      }
      // wave32 butterfly argmin (first-index tiebreak, matches jnp.argmin)
      for (int off = 16; off > 0; off >>= 1) {
        const float ob = __shfl_xor(best, off, 32);
        const int   oj = __shfl_xor(bestj, off, 32);
        if (ob < best || (ob == best && oj < bestj)) { best = ob; bestj = oj; }
      }
      __syncthreads();

      const float delta = best;
      for (int j = lane; j < M1; j += 32) {
        if (used[j]) { u[p[j]] += delta; v[j] -= delta; }  // p[] injective on used
        else         { minv[j] -= delta; }
      }
      if (lane == 0) sj0 = bestj;
      __syncthreads();
    }

    if (lane == 0) {  // augment along 'way' chain
      int j0 = sj0;
      while (j0 != 0) { const int j1 = way[j0]; p[j0] = p[j1]; j0 = j1; }
    }
    __syncthreads();
  }

  if (lane == 0) {  // exactly TT assigned columns; emit sorted by query index
    int k = 0;
    for (int q = 0; q < NQ && k < TT; ++q) {
      const int pi = p[q + 1];
      if (pi > 0) {
        out_rows[b * TT + k] = (float)q;
        out_cols[b * TT + k] = (float)(pi - 1);
        ++k;
      }
    }
  }
}

// ---------------------------------------------------------------------------
extern "C" void kernel_launch(void* const* d_in, const int* in_sizes, int n_in,
                              void* d_out, int out_size, void* d_ws, size_t ws_size,
                              hipStream_t stream) {
  const float* logits = (const float*)d_in[0];   // (64,300,2)
  const float* boxes  = (const float*)d_in[1];   // (64,300,4)
  const float* tgt    = (const float*)d_in[2];   // (1024,4)
  float* C        = (float*)d_out;               // 19200*1024
  float* out_rows = C + (size_t)NQT * NT;        // 64*16
  float* out_cols = out_rows + BS * TT;          // 64*16
  unsigned int* nprob_pk = (unsigned int*)d_ws;  // 19200 packed f16 pairs

  hm_prob_kernel<<<(NQT + 255) / 256, 256, 0, stream>>>(logits, nprob_pk);

  dim3 gridB(NQT / 16, NT / 128);  // 1200 x 8
  hm_cost_kernel<<<gridB, 256, 0, stream>>>(boxes, tgt, nprob_pk, C);

  hm_hungarian_kernel<<<BS, 32, 0, stream>>>(C, out_rows, out_cols);
}
